// GraphAttentionLayer_88742614270233
// MI455X (gfx1250) — matrix operand order repaired and checked by reading, not compiled, not run on hardware
//
#include <hip/hip_runtime.h>
#include <stdint.h>

#define NN 4096
#define CC 256
#define HH 8
#define HD 32

typedef __bf16        v16bf __attribute__((ext_vector_type(16)));
typedef float         v8f   __attribute__((ext_vector_type(8)));
typedef float         v16f  __attribute__((ext_vector_type(16)));
typedef unsigned int  v8u   __attribute__((ext_vector_type(8)));

// ---------------------------------------------------------------------------
// Kernel 1: projection GEMM  h[n,c] = sum_k X[n,k] * W[c,k] + b[c]
// bf16 WMMA 16x16x32; grid (N/16, C/64), 4 waves/block (one 16-col tile each)
// ---------------------------------------------------------------------------
__global__ __launch_bounds__(128) void k_proj(const float* __restrict__ X,
                                              const float* __restrict__ Ww,
                                              const float* __restrict__ Wb,
                                              float* __restrict__ h) {
  const int lane = threadIdx.x & 31;
  const int wave = threadIdx.x >> 5;
  const int lo = lane & 15, hi = lane >> 4;
  const int r0 = blockIdx.x * 16;
  const int c0 = blockIdx.y * 64 + wave * 16;
  const int arow = r0 + lo;     // A: M = lo
  const int bcol = c0 + lo;     // B: N = lo
  v8f acc = {};
  for (int k = 0; k < CC; k += 32) {
    // A 16x32 bf16: lane half hi -> K octets [hi*8, hi*8+8) and [16+hi*8, ...)
    const float* ap = X + arow * CC + k + hi * 8;
    v16f af;
#pragma unroll
    for (int e = 0; e < 8; ++e) af[e] = ap[e];
#pragma unroll
    for (int e = 0; e < 8; ++e) af[8 + e] = ap[16 + e];
    // B 32x16 bf16: col = lo, K octets [hi*8, hi*8+8) and [16+hi*8, ...) ; B[k,c] = W[c,k]
    const float* bp = Ww + bcol * CC + k + hi * 8;
    v16f bf;
#pragma unroll
    for (int e = 0; e < 16; ++e) bf[e] = bp[(e & 8) * 2 + (e & 7)];
    acc = __builtin_amdgcn_wmma_f32_16x16x32_bf16(
        false, __builtin_convertvector(af, v16bf),
        false, __builtin_convertvector(bf, v16bf),
        (short)0, acc, false, false);
  }
  const float bias = Wb[bcol];
#pragma unroll
  for (int r = 0; r < 8; ++r)
    h[(r0 + hi * 8 + r) * CC + bcol] = acc[r] + bias;
}

// ---------------------------------------------------------------------------
// Kernel 2a: per-(head,node) attention logits s_src / s_dst
// ---------------------------------------------------------------------------
__global__ __launch_bounds__(256) void k_scores(const float* __restrict__ h,
                                                const float* __restrict__ aw,
                                                float* __restrict__ s_src,
                                                float* __restrict__ s_dst) {
  const int t = blockIdx.x * 256 + threadIdx.x;
  const int n = t & (NN - 1);
  const int head = t >> 12;
  const float* hr = h + n * CC + head * HD;
  float sr = 0.f, sd = 0.f;
#pragma unroll
  for (int d = 0; d < HD; ++d) {
    const float v = hr[d];
    sr = fmaf(v, aw[d], sr);
    sd = fmaf(v, aw[HD + d], sd);
  }
  s_src[head * NN + n] = sr;
  s_dst[head * NN + n] = sd;
}

// ---------------------------------------------------------------------------
// Kernel 2b: h (f32, row-major [n][c]) -> hT (bf16, [c][n]) for WMMA B loads
// ---------------------------------------------------------------------------
__global__ __launch_bounds__(256) void k_tobf(const float* __restrict__ h,
                                              unsigned short* __restrict__ hT) {
  const int t = blockIdx.x * 256 + threadIdx.x;
  const int n = t & (NN - 1);
  const int c = t >> 12;
  const __bf16 b = (__bf16)h[n * CC + c];
  hT[c * NN + n] = __builtin_bit_cast(unsigned short, b);
}

// 8 masked leaky-relu scores for K-octet starting at jb (lane-local)
__device__ __forceinline__ void scores8(const float* __restrict__ sd,
                                        const int* __restrict__ ar,
                                        int jb, float ssrc, float res[8]) {
  const float4 sa = *(const float4*)(sd + jb);
  const float4 sb = *(const float4*)(sd + jb + 4);
  const int4 aa = *(const int4*)(ar + jb);
  const int4 ab = *(const int4*)(ar + jb + 4);
  const float s[8] = {sa.x, sa.y, sa.z, sa.w, sb.x, sb.y, sb.z, sb.w};
  const int a[8] = {aa.x, aa.y, aa.z, aa.w, ab.x, ab.y, ab.z, ab.w};
#pragma unroll
  for (int e = 0; e < 8; ++e) {
    float v = ssrc + s[e];
    v = fmaxf(v, 0.2f * v);                 // leaky_relu (0.2 < 1)
    res[e] = a[e] ? v : -1.0e9f;            // adjacency mask
  }
}

// ---------------------------------------------------------------------------
// Kernel 3: fused flash-attention aggregation + residual + ELU
// block = 16 rows x 8 heads (8 waves, one head each); wave-independent.
// Phase 1: row max (no exp). Phase 2: exp once, denom accumulated alongside
// P@V WMMA, divide at epilogue.
// ---------------------------------------------------------------------------
__global__ __launch_bounds__(256) void k_attn(const float* __restrict__ X,
                                              const int* __restrict__ adj,
                                              const unsigned short* __restrict__ hT,
                                              const float* __restrict__ s_src,
                                              const float* __restrict__ s_dst,
                                              const float* __restrict__ a_b,
                                              float* __restrict__ out) {
  const int lane = threadIdx.x & 31;
  const int head = threadIdx.x >> 5;
  const int lo = lane & 15, hi = lane >> 4;
  const int i0 = blockIdx.x * 16;
  const int row = i0 + lo;
  const float ssrc = s_src[head * NN + row] + a_b[0];
  const float* sd = s_dst + head * NN;
  const int* arow = adj + row * NN;
  const float L2E = 1.4426950408889634f;

  // ---- phase 1: running row max over all j ----
  float m = -1e30f;
  for (int jc = 0; jc < NN; jc += 32) {
    __builtin_prefetch(arow + jc + 32 + hi * 8, 0, 0);  // next adjacency tile
    float r8[8];
    scores8(sd, arow, jc + hi * 8, ssrc, r8);
#pragma unroll
    for (int e = 0; e < 8; ++e) m = fmaxf(m, r8[e]);
    scores8(sd, arow, jc + 16 + hi * 8, ssrc, r8);
#pragma unroll
    for (int e = 0; e < 8; ++e) m = fmaxf(m, r8[e]);
  }
  // combine lane-pair halves of the same row (SWAPX16)
  m = fmaxf(m, __int_as_float(
                   __builtin_amdgcn_ds_swizzle(__float_as_int(m), 0x401F)));

  // ---- phase 2: P = exp2((s-m)*log2e), denom l, O += P @ h_head ----
  float l = 0.f;
  v8f acc0 = {}, acc1 = {};
  const unsigned short* bbase = hT + (head * HD + lo) * NN;
  for (int jc = 0; jc < NN; jc += 32) {
    __builtin_prefetch(arow + jc + 32 + hi * 8, 0, 0);
    v16f pf;
#pragma unroll
    for (int seg = 0; seg < 2; ++seg) {
      float r8[8];
      scores8(sd, arow, jc + hi * 16 + seg * 8, ssrc, r8);
#pragma unroll
      for (int e = 0; e < 8; ++e) {
        const float p = __builtin_amdgcn_exp2f((r8[e] - m) * L2E);
        pf[seg * 8 + e] = p;
        l += p;
      }
    }
    // B tiles: col d = lo (+16), K = jc + hi*16 + e, contiguous bf16
    const v8u b0 = *(const v8u*)(bbase + jc + hi * 16);
    const v8u b1 = *(const v8u*)(bbase + 16 * NN + jc + hi * 16);
    const v16bf A = __builtin_convertvector(pf, v16bf);
    acc0 = __builtin_amdgcn_wmma_f32_16x16x32_bf16(
        false, A, false, __builtin_bit_cast(v16bf, b0), (short)0, acc0, false, false);
    acc1 = __builtin_amdgcn_wmma_f32_16x16x32_bf16(
        false, A, false, __builtin_bit_cast(v16bf, b1), (short)0, acc1, false, false);
  }
  // total denominator for row `lo` (both halves)
  float lrow = l + __int_as_float(
                       __builtin_amdgcn_ds_swizzle(__float_as_int(l), 0x401F));

  // ---- epilogue: normalize, residual, ELU ----
#pragma unroll
  for (int r = 0; r < 8; ++r) {
    const int orow = hi * 8 + r;  // C-tile row for this lane/VGPR
    const float lr = __int_as_float(
        __builtin_amdgcn_ds_bpermute(orow << 2, __float_as_int(lrow)));
    const float linv = __builtin_amdgcn_rcpf(lr);
    const int idx0 = (i0 + orow) * CC + head * HD + lo;
    float v0 = fmaf(acc0[r], linv, X[idx0]);
    out[idx0] = v0 > 0.f ? v0 : expm1f(v0);
    const int idx1 = idx0 + 16;
    float v1 = fmaf(acc1[r], linv, X[idx1]);
    out[idx1] = v1 > 0.f ? v1 : expm1f(v1);
  }
}

// ---------------------------------------------------------------------------
extern "C" void kernel_launch(void* const* d_in, const int* in_sizes, int n_in,
                              void* d_out, int out_size, void* d_ws, size_t ws_size,
                              hipStream_t stream) {
  const float* X   = (const float*)d_in[0];
  const int*   adj = (const int*)d_in[1];
  const float* Ww  = (const float*)d_in[2];
  const float* Wb  = (const float*)d_in[3];
  const float* aw  = (const float*)d_in[4];
  const float* ab  = (const float*)d_in[5];
  float* out = (float*)d_out;

  char* ws = (char*)d_ws;
  float* h = (float*)ws;                                            // 4 MB
  unsigned short* hT = (unsigned short*)(ws + (size_t)NN * CC * 4); // 2 MB
  float* ssrc = (float*)(ws + (size_t)NN * CC * 4 + (size_t)NN * CC * 2);
  float* sdst = ssrc + HH * NN;

  k_proj<<<dim3(NN / 16, CC / 64), 128, 0, stream>>>(X, Ww, Wb, h);
  k_scores<<<(NN * HH) / 256, 256, 0, stream>>>(h, aw, ssrc, sdst);
  k_tobf<<<(NN * CC) / 256, 256, 0, stream>>>(h, hT);
  k_attn<<<NN / 16, 256, 0, stream>>>(X, adj, hT, ssrc, sdst, ab, out);
}
